// ScaledDotProductAttention_23450521436217
// MI455X (gfx1250) — matrix operand order repaired and checked by reading, not compiled, not run on hardware
//
#include <hip/hip_runtime.h>
#include <stdint.h>

typedef __attribute__((ext_vector_type(16))) __bf16 v16bf;
typedef __attribute__((ext_vector_type(8)))  __bf16 v8bf;
typedef __attribute__((ext_vector_type(8)))  float  v8f;

#define S_LEN    8192
#define D_K      128
#define BM       64            // query rows per block (4 waves x 16)
#define BN       64            // keys per KV iteration
#define NWAVE    4
#define NTHREADS 128
#define KPAD     (D_K + 16)    // padded bf16 K-tile row stride: 288B, 32B-aligned
#define VPAD     (BN + 16)     // padded bf16 V^T row stride: 160B, 32B-aligned
#define PPAD     (BN + 16)     // padded P row stride

// 16 async f32 float4 copies per thread: tile (BN*D_K floats) mirrored
// global->LDS, INST_OFFSET applies to both global and LDS address.
__device__ __forceinline__ void async_copy_tile_f32(const float* __restrict__ gsrc,
                                                    uint32_t lds_byte_base, int tid) {
  const float* gp = gsrc + tid * 4;                 // 16B per lane per op
  uint32_t     lp = lds_byte_base + tid * 16;
  asm volatile(
      "global_load_async_to_lds_b128 %0, %1, off\n\t"
      "global_load_async_to_lds_b128 %0, %1, off offset:2048\n\t"
      "global_load_async_to_lds_b128 %0, %1, off offset:4096\n\t"
      "global_load_async_to_lds_b128 %0, %1, off offset:6144\n\t"
      "global_load_async_to_lds_b128 %0, %1, off offset:8192\n\t"
      "global_load_async_to_lds_b128 %0, %1, off offset:10240\n\t"
      "global_load_async_to_lds_b128 %0, %1, off offset:12288\n\t"
      "global_load_async_to_lds_b128 %0, %1, off offset:14336\n\t"
      "global_load_async_to_lds_b128 %0, %1, off offset:16384\n\t"
      "global_load_async_to_lds_b128 %0, %1, off offset:18432\n\t"
      "global_load_async_to_lds_b128 %0, %1, off offset:20480\n\t"
      "global_load_async_to_lds_b128 %0, %1, off offset:22528\n\t"
      "global_load_async_to_lds_b128 %0, %1, off offset:24576\n\t"
      "global_load_async_to_lds_b128 %0, %1, off offset:26624\n\t"
      "global_load_async_to_lds_b128 %0, %1, off offset:28672\n\t"
      "global_load_async_to_lds_b128 %0, %1, off offset:30720"
      :: "v"(lp), "v"(gp) : "memory");
}

__device__ __forceinline__ void wait_asynccnt0() {
  asm volatile("s_wait_asynccnt 0x0" ::: "memory");
}

__global__ __launch_bounds__(NTHREADS, 1)
void fattn_wmma_bf16_kernel(const float* __restrict__ Q,
                            const float* __restrict__ K,
                            const float* __restrict__ V,
                            float* __restrict__ O) {
  // f32 async landing buffers (raw copies of the global tiles)
  __shared__ alignas(16) float  Kf[BN * D_K];            // 32 KB
  __shared__ alignas(16) float  Vf[BN * D_K];            // 32 KB
  // bf16 compute tiles
  __shared__ alignas(32) __bf16 Ksh[BN * KPAD];          // [kv][d] row-major
  __shared__ alignas(32) __bf16 Vsh[D_K * VPAD];         // [d][kv] transposed
  __shared__ alignas(32) __bf16 Psh[NWAVE * 16 * PPAD];  // per-wave P scratch

  const int tid  = threadIdx.x;
  const int wave = tid >> 5;
  const int lane = tid & 31;
  const int l16  = lane & 15;
  const int hi   = lane >> 4;

  const int   qBase = blockIdx.x * BM + wave * 16;
  const float scale = 0.08838834764831845f;  // 1/sqrt(128)

  const uint32_t kfB = (uint32_t)(uintptr_t)&Kf[0];  // LDS_ADDR = generic[31:0]
  const uint32_t vfB = (uint32_t)(uintptr_t)&Vf[0];

  // ---- kick async copy of tile 0 before anything else ----
  async_copy_tile_f32(K, kfB, tid);
  async_copy_tile_f32(V, vfB, tid);

  // ---- Q tile -> bf16 A-fragments, kept in VGPRs for the whole KV loop ----
  // A operand, this kernel's contiguous k order: row M = l&15; K = 16*hi+0..15 (elems 0..15),
  // per 32-wide chunk.
  v16bf qfrag[4];
  {
    const float* qrow = Q + (size_t)(qBase + l16) * D_K;
    #pragma unroll
    for (int c = 0; c < 4; ++c) {
      const int d0 = c * 32 + hi * 16;
      #pragma unroll
      for (int j = 0; j < 8; ++j) {
        qfrag[c][j]     = (__bf16)qrow[d0 + j];
        qfrag[c][j + 8] = (__bf16)qrow[d0 + 8 + j];
      }
    }
  }

  v8f oacc[8];
  #pragma unroll
  for (int i = 0; i < 8; ++i) oacc[i] = (v8f){};
  float m_i[8], l_i[8];
  #pragma unroll
  for (int v = 0; v < 8; ++v) { m_i[v] = -3.0e38f; l_i[v] = 0.0f; }

  __bf16* pw = Psh + wave * (16 * PPAD);

  for (int kv0 = 0; kv0 < S_LEN; kv0 += BN) {
    // ---- tile arrived (own waves) + everyone done with previous bf16 tiles
    wait_asynccnt0();
    __syncthreads();

    // ---- convert pass: f32 landing buffer -> bf16 K (row-major) / V^T ----
    #pragma unroll
    for (int e = 0; e < (BN * D_K / 4) / NTHREADS; ++e) {  // 16 float4/thread
      const int i   = e * NTHREADS + tid;
      const int row = i >> 5;            // 32 float4 per 128-wide row
      const int dq  = (i & 31) << 2;
      const float4 kk = *(const float4*)&Kf[i * 4];
      Ksh[row * KPAD + dq + 0] = (__bf16)kk.x;
      Ksh[row * KPAD + dq + 1] = (__bf16)kk.y;
      Ksh[row * KPAD + dq + 2] = (__bf16)kk.z;
      Ksh[row * KPAD + dq + 3] = (__bf16)kk.w;
      const float4 vv = *(const float4*)&Vf[i * 4];
      Vsh[(dq + 0) * VPAD + row] = (__bf16)vv.x;
      Vsh[(dq + 1) * VPAD + row] = (__bf16)vv.y;
      Vsh[(dq + 2) * VPAD + row] = (__bf16)vv.z;
      Vsh[(dq + 3) * VPAD + row] = (__bf16)vv.w;
    }
    __syncthreads();  // landing buffer drained, bf16 tiles ready

    // ---- overlap: kick async copy of the NEXT tile during compute ----
    if (kv0 + BN < S_LEN) {
      async_copy_tile_f32(K + (size_t)(kv0 + BN) * D_K, kfB, tid);
      async_copy_tile_f32(V + (size_t)(kv0 + BN) * D_K, vfB, tid);
    }

    // ---- S = Q * K^T (16 x 64 per wave): batch 8 ds_loads, then 4 WMMAs ----
    // B operand, same contiguous k order as A: lane holds col N = l&15; K = 16*hi + 0..15.
    v8f sacc[4];
    #pragma unroll
    for (int nb = 0; nb < 4; ++nb) {
      v16bf bfr[4];
      #pragma unroll
      for (int c = 0; c < 4; ++c)
        bfr[c] = *(const v16bf*)(Ksh + (nb * 16 + l16) * KPAD + c * 32 + hi * 16);
      v8f acc = (v8f){};
      #pragma unroll
      for (int c = 0; c < 4; ++c)
        acc = __builtin_amdgcn_wmma_f32_16x16x32_bf16(
            false, qfrag[c], false, bfr[c], (short)0, acc, false, false);
      sacc[nb] = acc;
    }

    // ---- online softmax (C/D layout: row m = v + 8*hi, col n = l&15) ----
    float pmax[8];
    #pragma unroll
    for (int v = 0; v < 8; ++v) {
      float m = fmaxf(fmaxf(sacc[0][v], sacc[1][v]),
                      fmaxf(sacc[2][v], sacc[3][v])) * scale;
      m = fmaxf(m, __shfl_xor(m, 1, 32));
      m = fmaxf(m, __shfl_xor(m, 2, 32));
      m = fmaxf(m, __shfl_xor(m, 4, 32));
      m = fmaxf(m, __shfl_xor(m, 8, 32));
      pmax[v] = m;
    }
    float alpha[8], psum[8];
    #pragma unroll
    for (int v = 0; v < 8; ++v) {
      const float mnew = fmaxf(m_i[v], pmax[v]);
      alpha[v] = __expf(m_i[v] - mnew);
      m_i[v]   = mnew;
      psum[v]  = 0.0f;
    }
    #pragma unroll
    for (int nb = 0; nb < 4; ++nb) {
      #pragma unroll
      for (int v = 0; v < 8; ++v) {
        const float p = __expf(sacc[nb][v] * scale - m_i[v]);
        sacc[nb][v] = p;
        psum[v] += p;
      }
    }
    #pragma unroll
    for (int v = 0; v < 8; ++v) {
      float s = psum[v];
      s += __shfl_xor(s, 1, 32);
      s += __shfl_xor(s, 2, 32);
      s += __shfl_xor(s, 4, 32);
      s += __shfl_xor(s, 8, 32);
      l_i[v] = l_i[v] * alpha[v] + s;
    }
    #pragma unroll
    for (int nb2 = 0; nb2 < 8; ++nb2)
      #pragma unroll
      for (int v = 0; v < 8; ++v)
        oacc[nb2][v] *= alpha[v];

    // ---- P: C-layout -> bf16 in per-wave LDS -> reload in A-layout ----
    // Same-wave DS ops are DScnt-ordered: no barrier needed.
    #pragma unroll
    for (int nb = 0; nb < 4; ++nb)
      #pragma unroll
      for (int v = 0; v < 8; ++v)
        pw[(v + 8 * hi) * PPAD + nb * 16 + l16] = (__bf16)sacc[nb][v];

    // ---- O += P * V (16 x 128 per wave): batch loads, then WMMA chains ----
    #pragma unroll
    for (int c = 0; c < 2; ++c) {
      const v8bf* ap = (const v8bf*)(pw + l16 * PPAD + c * 32 + hi * 16);
      const v8bf lo  = ap[0];
      const v8bf hi8 = ap[1];  // +16B = +8 bf16 elems
      v16bf afrag;
      #pragma unroll
      for (int j = 0; j < 8; ++j) { afrag[j] = lo[j]; afrag[j + 8] = hi8[j]; }
      #pragma unroll
      for (int g = 0; g < 2; ++g) {       // two groups of 4 d-blocks
        v16bf vfr[4];
        #pragma unroll
        for (int t = 0; t < 4; ++t)
          vfr[t] = *(const v16bf*)(Vsh + ((g * 4 + t) * 16 + l16) * VPAD +
                                   c * 32 + hi * 16);
        #pragma unroll
        for (int t = 0; t < 4; ++t)
          oacc[g * 4 + t] = __builtin_amdgcn_wmma_f32_16x16x32_bf16(
              false, afrag, false, vfr[t], (short)0, oacc[g * 4 + t],
              false, false);
      }
    }
  }

  // ---- epilogue: normalize by softmax sum, store fp32 ----
  float inv_l[8];
  #pragma unroll
  for (int v = 0; v < 8; ++v) inv_l[v] = 1.0f / l_i[v];
  float* orow = O + (size_t)qBase * D_K;
  #pragma unroll
  for (int nb2 = 0; nb2 < 8; ++nb2)
    #pragma unroll
    for (int v = 0; v < 8; ++v)
      orow[(size_t)(v + 8 * hi) * D_K + nb2 * 16 + l16] = oacc[nb2][v] * inv_l[v];
}

extern "C" void kernel_launch(void* const* d_in, const int* in_sizes, int n_in,
                              void* d_out, int out_size, void* d_ws, size_t ws_size,
                              hipStream_t stream) {
  const float* Q = (const float*)d_in[0];
  const float* K = (const float*)d_in[1];
  const float* V = (const float*)d_in[2];
  float* O = (float*)d_out;
  (void)in_sizes; (void)n_in; (void)out_size; (void)d_ws; (void)ws_size;
  dim3 grid(S_LEN / BM);
  dim3 block(NTHREADS);
  hipLaunchKernelGGL(fattn_wmma_bf16_kernel, grid, block, 0, stream, Q, K, V, O);
}
